// DecoderWithAttention_68882685493806
// MI455X (gfx1250) — compile-verified, matrix operand order checked by reading, not run on hardware
//
#include <hip/hip_runtime.h>
#include <hip/hip_bf16.h>

// ---------------------------------------------------------------------------
// Show-Attend-Tell decoder for MI455X (gfx1250, wave32, WMMA bf16 path).
// Heavy GEMMs (ea precompute, fused LSTM gates, vocab fc) use
// v_wmma_f32_16x16x32_bf16 with bf16 weights converted once per launch.
// Each wave owns a 64(M)x16(N) strip; fragments are double-buffered across
// K-chunks so WMMAs overlap with in-flight global loads (no loadcnt-0 drains).
// ---------------------------------------------------------------------------

#define BATCH   64
#define PS      14
#define P       (PS * PS)      // 196
#define ENC     512
#define DIM     512
#define ATT     512
#define EMB     512
#define VOCAB   32000
#define MAXLEN  51
#define TSTEPS  (MAXLEN - 1)   // 50
#define KCAT    (EMB + ENC + DIM) // 1536: [emb | ctx | h]
#define NGATE   (4 * DIM)      // 2048

typedef __attribute__((ext_vector_type(16))) __bf16 v16bf;
typedef __attribute__((ext_vector_type(8)))  float  v8f;

__device__ __forceinline__ __bf16 to_bf16(float f) {
  union { float f; unsigned u; } x; x.f = f;
  unsigned r = x.u + 0x7FFFu + ((x.u >> 16) & 1u);   // round-to-nearest-even
  unsigned short h = (unsigned short)(r >> 16);
  return __builtin_bit_cast(__bf16, h);
}

__device__ __forceinline__ float sigmoidf(float x) {
  return 1.0f / (1.0f + expf(-x));
}

__device__ __forceinline__ v8f wmma_bf16(v16bf a, v16bf b, v8f c) {
  return __builtin_amdgcn_wmma_f32_16x16x32_bf16(
      /*neg_a=*/false, a, /*neg_b=*/false, b,
      /*c_mod=*/(short)0, c, /*reuse_a=*/false, /*reuse_b=*/false);
}

// One K-chunk's worth of fragments: B tile + 4 A tiles (64 M rows).
struct Frag {
  v16bf b, a0, a1, a2, a3;
};

__device__ __forceinline__ void load_frag(Frag& f,
                                          const __bf16* __restrict__ wRow,
                                          const __bf16* __restrict__ aBase,
                                          long aS, int k) {
  f.b  = *(const v16bf*)(wRow + k);
  f.a0 = *(const v16bf*)(aBase + k);
  f.a1 = *(const v16bf*)(aBase + aS + k);
  f.a2 = *(const v16bf*)(aBase + 2 * aS + k);
  f.a3 = *(const v16bf*)(aBase + 3 * aS + k);
}

// ---------------------------------------------------------------------------
// WMMA GEMM: out[m, n] = sum_k A[m,k] * W[n,k] (+ bias[n]), bf16 in, f32 out.
// A: (M, K) row-major bf16 (M multiple of 64). W: (N, K) row-major bf16
// (N multiple of 128, K multiple of 64) -> computes A @ W^T.
// Wave tile: 64(M) x 16(N); B fragment reused across 4 independent
// accumulator chains. Block = 8 waves = 64(M) x 128(N).
// Optional fc masking fused into the store.
// ---------------------------------------------------------------------------
__global__ void gemm_bf16_wmma(const __bf16* __restrict__ A,
                               const __bf16* __restrict__ W,
                               const float* __restrict__ bias,
                               float* __restrict__ out,
                               int M, int N, int K, long out_stride,
                               const int* __restrict__ dec_len, int t) {
  const int wave = threadIdx.x >> 5;
  const int lane = threadIdx.x & 31;
  const int m0 = blockIdx.x * 64;
  const int n0 = blockIdx.y * 128 + wave * 16;
  if (m0 >= M || n0 >= N) return;               // uniform per wave: EXEC all-1s

  const int half = lane >> 4;                   // 0: K 0..15, 1: K 16..31
  const int r    = lane & 15;
  const __bf16* aBase = A + (long)(m0 + r) * K + half * 16;
  const __bf16* wRow  = W + (long)(n0 + r) * K + half * 16;
  const long aS = (long)16 * K;                 // 16-row stride in A

  v8f acc0 = {}, acc1 = {}, acc2 = {}, acc3 = {};

  // Double-buffered K pipeline: loads for the next chunk are in flight while
  // the current chunk's 4 WMMAs execute.
  Frag f0, f1;
  load_frag(f0, wRow, aBase, aS, 0);
  for (int k = 0; k < K; k += 64) {
    __builtin_prefetch(wRow + k + 256, 0, 1);   // global_prefetch_b8 ahead
    load_frag(f1, wRow, aBase, aS, k + 32);
    acc0 = wmma_bf16(f0.a0, f0.b, acc0);
    acc1 = wmma_bf16(f0.a1, f0.b, acc1);
    acc2 = wmma_bf16(f0.a2, f0.b, acc2);
    acc3 = wmma_bf16(f0.a3, f0.b, acc3);
    if (k + 64 < K) load_frag(f0, wRow, aBase, aS, k + 64);
    acc0 = wmma_bf16(f1.a0, f1.b, acc0);
    acc1 = wmma_bf16(f1.a1, f1.b, acc1);
    acc2 = wmma_bf16(f1.a2, f1.b, acc2);
    acc3 = wmma_bf16(f1.a3, f1.b, acc3);
  }

  const int n = n0 + r;
  const float bi = bias ? bias[n] : 0.0f;
  v8f accs[4] = {acc0, acc1, acc2, acc3};
#pragma unroll
  for (int mt = 0; mt < 4; ++mt) {
#pragma unroll
    for (int v = 0; v < 8; ++v) {
      const int m = m0 + mt * 16 + v + half * 8;  // ISA C/D layout
      float val = accs[mt][v] + bi;
      if (dec_len) val = (t < dec_len[m]) ? val : 0.0f;
      out[(long)m * out_stride + n] = val;
    }
  }
}

// ---------------------------------------------------------------------------
// Preprocessing kernels
// ---------------------------------------------------------------------------

// Stable descending argsort of cap_len (B=64) via rank counting; one block.
__global__ void sort_kernel(const int* __restrict__ cap_len,
                            int* __restrict__ order, int* __restrict__ dec_len,
                            float* __restrict__ out_declen,
                            float* __restrict__ out_order) {
  __shared__ int keys[BATCH];
  const int b = threadIdx.x;
  keys[b] = cap_len[b];
  __syncthreads();
  const int my = keys[b];
  int pos = 0;
  for (int j = 0; j < BATCH; ++j) {
    const int kj = keys[j];
    if (kj > my || (kj == my && j < b)) ++pos;  // stable descending rank
  }
  order[pos]      = b;
  dec_len[pos]    = my - 1;
  out_declen[pos] = (float)(my - 1);
  out_order[pos]  = (float)b;
}

__global__ void gather_caps(const int* __restrict__ caps,
                            const int* __restrict__ order,
                            int* __restrict__ caps_s,
                            float* __restrict__ out_caps) {
  const int i = blockIdx.x;
  const int src = order[i];
  for (int t = threadIdx.x; t < MAXLEN; t += blockDim.x) {
    const int v = caps[src * MAXLEN + t];
    caps_s[i * MAXLEN + t]  = v;
    out_caps[i * MAXLEN + t] = (float)v;
  }
}

// Gather encoder features by sorted order; emit f32 + bf16 copies and mean.
__global__ void gather_enc(const float* __restrict__ enc_in,
                           const int* __restrict__ order,
                           float* __restrict__ enc_f,
                           __bf16* __restrict__ enc_b,
                           float* __restrict__ mean_enc) {
  const int b = blockIdx.x;
  const int e = blockIdx.y * blockDim.x + threadIdx.x;   // 0..511
  const float* src = enc_in + (long)order[b] * P * ENC + e;
  float s = 0.0f;
  for (int p = 0; p < P; ++p) {
    const float v = src[(long)p * ENC];
    const long dst = ((long)b * P + p) * ENC + e;
    enc_f[dst] = v;
    enc_b[dst] = to_bf16(v);
    s += v;
  }
  mean_enc[b * ENC + e] = s * (1.0f / (float)P);
}

__global__ void convert_bf16(const float* __restrict__ src,
                             __bf16* __restrict__ dst, long n) {
  const long stride = (long)gridDim.x * blockDim.x;
  for (long i = (long)blockIdx.x * blockDim.x + threadIdx.x; i < n; i += stride)
    dst[i] = to_bf16(src[i]);
}

// Fuse W_ih (2048,1024) and W_hh (2048,512) into Wcat (2048,1536) bf16.
__global__ void build_wcat(const float* __restrict__ W_ih,
                           const float* __restrict__ W_hh,
                           __bf16* __restrict__ Wcat) {
  const long stride = (long)gridDim.x * blockDim.x;
  const long n = (long)NGATE * KCAT;
  for (long i = (long)blockIdx.x * blockDim.x + threadIdx.x; i < n; i += stride) {
    const long row = i / KCAT, k = i % KCAT;
    const float v = (k < (EMB + ENC)) ? W_ih[row * (EMB + ENC) + k]
                                      : W_hh[row * DIM + (k - (EMB + ENC))];
    Wcat[i] = to_bf16(v);
  }
}

__global__ void bias_sum_k(const float* __restrict__ a,
                           const float* __restrict__ b,
                           float* __restrict__ out, int n) {
  const int i = blockIdx.x * blockDim.x + threadIdx.x;
  if (i < n) out[i] = a[i] + b[i];
}

// h0 / c0 from mean encoder features (small f32 GEMV pair).
__global__ void init_hc(const float* __restrict__ mean_enc,
                        const float* __restrict__ Wh, const float* __restrict__ bh,
                        const float* __restrict__ Wc, const float* __restrict__ bc,
                        float* __restrict__ h, float* __restrict__ c) {
  const int idx = blockIdx.x * blockDim.x + threadIdx.x;   // 64*512
  const int b = idx >> 9, n = idx & 511;
  const float* m = mean_enc + b * ENC;
  float sh = bh[n], sc = bc[n];
  for (int k = 0; k < ENC; ++k) {
    const float mv = m[k];
    sh = fmaf(mv, Wh[n * ENC + k], sh);
    sc = fmaf(mv, Wc[n * ENC + k], sc);
  }
  h[idx] = sh;
  c[idx] = sc;
}

// ---------------------------------------------------------------------------
// Per-timestep kernels
// ---------------------------------------------------------------------------

// da = h @ W_dec^T + b; gate = sigmoid(h @ W_fb^T + b); stash bf16(h) into xh.
__global__ void attn_pre(const float* __restrict__ h,
                         const float* __restrict__ W_dec, const float* __restrict__ b_dec,
                         const float* __restrict__ W_fb, const float* __restrict__ b_fb,
                         float* __restrict__ da, float* __restrict__ gate,
                         __bf16* __restrict__ xh) {
  const int idx = blockIdx.x * blockDim.x + threadIdx.x;   // 64*512
  const int b = idx >> 9, n = idx & 511;
  const float* hr = h + b * DIM;
  float s1 = b_dec[n], s2 = b_fb[n];
  for (int k = 0; k < DIM; ++k) {
    const float hv = hr[k];
    s1 = fmaf(hv, W_dec[n * DIM + k], s1);
    s2 = fmaf(hv, W_fb[n * DIM + k], s2);
  }
  da[idx]   = s1;
  gate[idx] = sigmoidf(s2);
  xh[(long)b * KCAT + (EMB + ENC) + n] = to_bf16(hr[n]);   // h slice of [x|h]
}

// scores[b,p] = sum_a relu(ea[b,p,a] + da[b,a]) * w_full[a] + b_full. Wave/(b,p).
__global__ void attn_scores(const float* __restrict__ ea,
                            const float* __restrict__ da,
                            const float* __restrict__ wfull,
                            const float* __restrict__ bfull,
                            float* __restrict__ scores) {
  const int wid  = (blockIdx.x * blockDim.x + threadIdx.x) >> 5;
  const int lane = threadIdx.x & 31;
  if (wid >= BATCH * P) return;
  const int b = wid / P;
  const float* er = ea + (long)wid * ATT;
  const float* dr = da + b * ATT;
  float s = 0.0f;
  for (int k = lane; k < ATT; k += 32) {
    float v = er[k] + dr[k];
    v = v > 0.0f ? v : 0.0f;
    s = fmaf(v, wfull[k], s);
  }
  for (int off = 16; off > 0; off >>= 1) s += __shfl_down(s, off, 32);
  if (lane == 0) scores[wid] = s + bfull[0];
}

// Softmax over P=196 per batch row; masked alpha straight into d_out.
__global__ void attn_softmax(const float* __restrict__ scores,
                             float* __restrict__ alpha,
                             float* __restrict__ out_alphas,
                             const int* __restrict__ dec_len, int t) {
  __shared__ float sm[256];
  const int b  = blockIdx.x;
  const int p  = threadIdx.x;
  const float v = (p < P) ? scores[b * P + p] : -3.4e38f;
  sm[p] = v; __syncthreads();
  for (int s = 128; s > 0; s >>= 1) {
    if (p < s) sm[p] = fmaxf(sm[p], sm[p + s]);
    __syncthreads();
  }
  const float mx = sm[0]; __syncthreads();
  const float e = (p < P) ? expf(v - mx) : 0.0f;
  sm[p] = e; __syncthreads();
  for (int s = 128; s > 0; s >>= 1) {
    if (p < s) sm[p] += sm[p + s];
    __syncthreads();
  }
  const float inv = 1.0f / sm[0];
  if (p < P) {
    const float a = e * inv;
    alpha[b * P + p] = a;
    const float m = (t < dec_len[b]) ? 1.0f : 0.0f;
    out_alphas[((long)b * TSTEPS + t) * P + p] = a * m;
  }
}

// ctx = gate * (alpha . enc); assemble bf16 x = [emb_t | ctx] into xh.
__global__ void attn_ctx(const float* __restrict__ alpha,
                         const float* __restrict__ enc_f,
                         const float* __restrict__ gate,
                         const float* __restrict__ emb_table,
                         const int* __restrict__ caps_s, int t,
                         __bf16* __restrict__ xh) {
  const int idx = blockIdx.x * blockDim.x + threadIdx.x;   // 64*512
  const int b = idx >> 9, e = idx & 511;
  const float* ar = alpha + b * P;
  const float* er = enc_f + (long)b * P * ENC + e;
  float s = 0.0f;
  for (int p = 0; p < P; ++p) s = fmaf(ar[p], er[(long)p * ENC], s);
  s *= gate[idx];
  const int tok = caps_s[b * MAXLEN + t];
  xh[(long)b * KCAT + e]       = to_bf16(emb_table[(long)tok * EMB + e]);
  xh[(long)b * KCAT + EMB + e] = to_bf16(s);
}

// LSTM pointwise: gates g (64,2048) -> h,c update; emit bf16(h) for fc GEMM.
__global__ void lstm_cell(const float* __restrict__ g,
                          float* __restrict__ h, float* __restrict__ c,
                          __bf16* __restrict__ h_b) {
  const int idx = blockIdx.x * blockDim.x + threadIdx.x;   // 64*512
  const int b = idx >> 9, d = idx & 511;
  const float* gr = g + b * NGATE;
  const float gi = sigmoidf(gr[d]);
  const float gf = sigmoidf(gr[DIM + d]);
  const float gg = tanhf(gr[2 * DIM + d]);
  const float go = sigmoidf(gr[3 * DIM + d]);
  const float cn = gf * c[idx] + gi * gg;
  const float hn = go * tanhf(cn);
  c[idx]   = cn;
  h[idx]   = hn;
  h_b[idx] = to_bf16(hn);
}

// ---------------------------------------------------------------------------
// Host launcher
// ---------------------------------------------------------------------------
static inline size_t alignUp(size_t x) { return (x + 255) & ~(size_t)255; }

extern "C" void kernel_launch(void* const* d_in, const int* in_sizes, int n_in,
                              void* d_out, int out_size, void* d_ws, size_t ws_size,
                              hipStream_t stream) {
  (void)in_sizes; (void)n_in; (void)out_size; (void)ws_size;

  const float* enc_in   = (const float*)d_in[0];
  const int*   caps_in  = (const int*)  d_in[1];
  const int*   cap_len  = (const int*)  d_in[2];
  const float* W_enc    = (const float*)d_in[3];
  const float* b_enc    = (const float*)d_in[4];
  const float* W_dec    = (const float*)d_in[5];
  const float* b_dec    = (const float*)d_in[6];
  const float* w_full   = (const float*)d_in[7];
  const float* b_full   = (const float*)d_in[8];
  const float* emb_tab  = (const float*)d_in[9];
  const float* W_ih     = (const float*)d_in[10];
  const float* W_hh     = (const float*)d_in[11];
  const float* b_ih     = (const float*)d_in[12];
  const float* b_hh     = (const float*)d_in[13];
  const float* W_init_h = (const float*)d_in[14];
  const float* b_init_h = (const float*)d_in[15];
  const float* W_init_c = (const float*)d_in[16];
  const float* b_init_c = (const float*)d_in[17];
  const float* W_fb     = (const float*)d_in[18];
  const float* b_fb     = (const float*)d_in[19];
  const float* W_fc     = (const float*)d_in[20];
  const float* b_fc     = (const float*)d_in[21];

  // d_out layout (floats): preds | caps | dec_len | alphas | order
  float* out_preds  = (float*)d_out;
  float* out_caps   = out_preds  + (size_t)BATCH * TSTEPS * VOCAB;
  float* out_declen = out_caps   + (size_t)BATCH * MAXLEN;
  float* out_alphas = out_declen + BATCH;
  float* out_order  = out_alphas + (size_t)BATCH * TSTEPS * P;

  // Workspace carve-out (256B aligned)
  char* base = (char*)d_ws;
  size_t off = 0;
  auto carve = [&](size_t bytes) { void* p = base + off; off = alignUp(off + bytes); return p; };

  int*    ws_order  = (int*)   carve(BATCH * sizeof(int));
  int*    ws_declen = (int*)   carve(BATCH * sizeof(int));
  int*    ws_caps   = (int*)   carve((size_t)BATCH * MAXLEN * sizeof(int));
  float*  ws_encf   = (float*) carve((size_t)BATCH * P * ENC * sizeof(float));
  __bf16* ws_encb   = (__bf16*)carve((size_t)BATCH * P * ENC * sizeof(__bf16));
  float*  ws_mean   = (float*) carve((size_t)BATCH * ENC * sizeof(float));
  float*  ws_ea     = (float*) carve((size_t)BATCH * P * ATT * sizeof(float));
  __bf16* ws_Wencb  = (__bf16*)carve((size_t)ATT * ENC * sizeof(__bf16));
  __bf16* ws_Wcatb  = (__bf16*)carve((size_t)NGATE * KCAT * sizeof(__bf16));
  __bf16* ws_Wfcb   = (__bf16*)carve((size_t)VOCAB * DIM * sizeof(__bf16));
  float*  ws_bsum   = (float*) carve(NGATE * sizeof(float));
  float*  ws_h      = (float*) carve((size_t)BATCH * DIM * sizeof(float));
  float*  ws_c      = (float*) carve((size_t)BATCH * DIM * sizeof(float));
  __bf16* ws_hb     = (__bf16*)carve((size_t)BATCH * DIM * sizeof(__bf16));
  float*  ws_da     = (float*) carve((size_t)BATCH * ATT * sizeof(float));
  float*  ws_gate   = (float*) carve((size_t)BATCH * ENC * sizeof(float));
  float*  ws_scores = (float*) carve((size_t)BATCH * P * sizeof(float));
  float*  ws_alpha  = (float*) carve((size_t)BATCH * P * sizeof(float));
  __bf16* ws_xh     = (__bf16*)carve((size_t)BATCH * KCAT * sizeof(__bf16));
  float*  ws_g      = (float*) carve((size_t)BATCH * NGATE * sizeof(float));

  // ---------------- Preprocessing (once per launch) ----------------
  sort_kernel<<<1, BATCH, 0, stream>>>(cap_len, ws_order, ws_declen,
                                       out_declen, out_order);
  gather_caps<<<BATCH, 64, 0, stream>>>(caps_in, ws_order, ws_caps, out_caps);
  gather_enc<<<dim3(BATCH, ENC / 256), 256, 0, stream>>>(enc_in, ws_order,
                                                         ws_encf, ws_encb, ws_mean);
  convert_bf16<<<512, 256, 0, stream>>>(W_enc, ws_Wencb, (long)ATT * ENC);
  convert_bf16<<<2048, 256, 0, stream>>>(W_fc, ws_Wfcb, (long)VOCAB * DIM);
  build_wcat<<<1024, 256, 0, stream>>>(W_ih, W_hh, ws_Wcatb);
  bias_sum_k<<<NGATE / 256, 256, 0, stream>>>(b_ih, b_hh, ws_bsum, NGATE);
  init_hc<<<(BATCH * DIM) / 256, 256, 0, stream>>>(ws_mean, W_init_h, b_init_h,
                                                   W_init_c, b_init_c, ws_h, ws_c);
  // ea = enc @ W_enc^T + b_enc : (12544, 512) = (12544,512) @ (512,512)^T
  gemm_bf16_wmma<<<dim3((BATCH * P) / 64, ATT / 128), 256, 0, stream>>>(
      ws_encb, ws_Wencb, b_enc, ws_ea, BATCH * P, ATT, ENC, ATT, nullptr, 0);

  // ---------------- Sequential decode: 50 dependent steps ----------------
  for (int t = 0; t < TSTEPS; ++t) {
    attn_pre<<<(BATCH * ATT) / 256, 256, 0, stream>>>(
        ws_h, W_dec, b_dec, W_fb, b_fb, ws_da, ws_gate, ws_xh);
    attn_scores<<<(BATCH * P * 32 + 255) / 256, 256, 0, stream>>>(
        ws_ea, ws_da, w_full, b_full, ws_scores);
    attn_softmax<<<BATCH, 256, 0, stream>>>(ws_scores, ws_alpha, out_alphas,
                                            ws_declen, t);
    attn_ctx<<<(BATCH * ENC) / 256, 256, 0, stream>>>(
        ws_alpha, ws_encf, ws_gate, emb_tab, ws_caps, t, ws_xh);
    // gates = [x|h] @ [W_ih|W_hh]^T + (b_ih + b_hh) : (64,2048), K=1536
    gemm_bf16_wmma<<<dim3(1, NGATE / 128), 256, 0, stream>>>(
        ws_xh, ws_Wcatb, ws_bsum, ws_g, BATCH, NGATE, KCAT, NGATE, nullptr, 0);
    lstm_cell<<<(BATCH * DIM) / 256, 256, 0, stream>>>(ws_g, ws_h, ws_c, ws_hb);
    // preds[:, t, :] = mask * (h_new @ W_fc^T + b_fc) : (64, 32000), K=512
    gemm_bf16_wmma<<<dim3(1, VOCAB / 128), 256, 0, stream>>>(
        ws_hb, ws_Wfcb, b_fc, out_preds + (long)t * VOCAB,
        BATCH, VOCAB, DIM, (long)TSTEPS * VOCAB, ws_declen, t);
  }
}
